// encoder_64398739637012
// MI455X (gfx1250) — compile-verified, matrix operand order checked by reading, not run on hardware
//
#include <hip/hip_runtime.h>
#include <math.h>

typedef __attribute__((ext_vector_type(16))) __bf16 v16bf;
typedef __attribute__((ext_vector_type(8)))  float  v8f;
typedef __attribute__((ext_vector_type(2)))  float  v2f;
typedef __attribute__((ext_vector_type(4)))  unsigned int u32x4;
typedef __attribute__((ext_vector_type(8)))  int i32x8;
typedef __attribute__((ext_vector_type(4)))  int i32x4;
typedef unsigned short u16;
typedef unsigned int   u32;

#define NSEQ   8192
#define DMODEL 512
#define NHEAD  8
#define DHEAD  64
#define NLAND  256
#define ATTD   256
#define CONVK  33

// ---------------- CDNA5 feature detection (per-pass; host pass falls back) ----
#if defined(__AMDGCN__) && __has_builtin(__builtin_amdgcn_global_load_async_to_lds_b128)
#define HAS_ASYNC_LDS 1
#else
#define HAS_ASYNC_LDS 0
#endif
#if defined(__AMDGCN__) && __has_builtin(__builtin_amdgcn_tensor_load_to_lds)
#define HAS_TDM 1
#else
#define HAS_TDM 0
#endif

#if HAS_ASYNC_LDS
__device__ __forceinline__ void async_ldg_b128(const void* g, void* l) {
  __builtin_amdgcn_global_load_async_to_lds_b128(
      (__attribute__((address_space(1))) i32x4*)g,
      (__attribute__((address_space(3))) i32x4*)l, 0, 0);
}
__device__ __forceinline__ void wait_async0() {
#if __has_builtin(__builtin_amdgcn_s_wait_asynccnt)
  __builtin_amdgcn_s_wait_asynccnt(0);
#else
  asm volatile("s_wait_asynccnt 0" ::: "memory");
#endif
}
#endif

#if HAS_TDM
__device__ __forceinline__ void wait_tensor0() {
#if __has_builtin(__builtin_amdgcn_s_wait_tensorcnt)
  __builtin_amdgcn_s_wait_tensorcnt(0);
#else
  asm volatile("s_wait_tensorcnt 0" ::: "memory");
#endif
}
// 2D tile load via Tensor Data Mover. data_size = 2 bytes (bf16).
// tile0 = contiguous elements per row, tile1 = rows, stride0 = row stride (elems).
// padIntervalCode/padAmountCode per D# group1 encoding (-1 disables LDS padding).
__device__ __forceinline__ void tdm_load_2d_bf16(u32 ldsAddr, const void* gaddr,
                                                 int tile0, int tile1, long stride0,
                                                 int padIntervalCode, int padAmountCode) {
  unsigned long long ga = (unsigned long long)gaddr;
  u32x4 g0;
  g0[0] = 1u;                                   // count=1 (valid user descriptor)
  g0[1] = ldsAddr;                              // lds_addr [63:32]
  g0[2] = (u32)ga;                              // global_addr low
  g0[3] = (u32)((ga >> 32) & 0x1FFFFFFu) | (2u << 30);  // addr[56:32] | type=2
  i32x8 g1;
  u32 flags = (1u << 16);                       // data_size = 1 -> 2 bytes
  if (padAmountCode >= 0)
    flags |= (1u << 20) | ((u32)padIntervalCode << 22) | ((u32)padAmountCode << 25);
  u32 dim0 = 1u << 20, dim1 = 1u << 20;         // large extents: no OOB clipping
  g1[0] = (int)flags;
  g1[1] = (int)((dim0 & 0xFFFFu) << 16);                        // tensor_dim0 lo16
  g1[2] = (int)(((dim0 >> 16) & 0xFFFFu) | ((dim1 & 0xFFFFu) << 16));
  g1[3] = (int)(((dim1 >> 16) & 0xFFFFu) | ((u32)tile0 << 16)); // tile_dim0
  g1[4] = (int)((u32)tile1 & 0xFFFFu);                          // tile_dim1
  g1[5] = (int)(u32)stride0;                                    // tensor_dim0_stride lo32
  g1[6] = (int)((u32)(stride0 >> 32) & 0xFFFFu);
  g1[7] = 0;
  i32x4 z4 = {};
  i32x8 z8 = {};
  __builtin_amdgcn_tensor_load_to_lds(g0, g1, z4, z4, z8, 0);
}
#endif

// ---------- helpers ----------
__device__ __forceinline__ u16 f2b(float f) {
  u32 u = __float_as_uint(f);
  u32 r = u + 0x7FFFu + ((u >> 16) & 1u);
  return (u16)(r >> 16);
}

union Frag16 { v16bf v; u16 u[16]; };

// ---------- elementwise kernels ----------
__global__ void k_cvt_bf16(const float* __restrict__ in, u16* __restrict__ out, long n) {
  long i = (long)blockIdx.x * blockDim.x + threadIdx.x;
  if (i < n) out[i] = f2b(in[i]);
}

__global__ void k_scale_q(float* __restrict__ qkv, float s, long n) {
  long i = (long)blockIdx.x * blockDim.x + threadIdx.x;
  if (i < n) {
    long r = i >> 9, c = i & 511;
    qkv[r * 1536 + c] *= s;
  }
}

__global__ void k_landmark_mean(const float* __restrict__ qkv,
                                float* __restrict__ ql, float* __restrict__ kl) {
  long i = (long)blockIdx.x * blockDim.x + threadIdx.x;
  if (i >= (long)NHEAD * NLAND * DHEAD) return;
  int d = i & 63, m = (i >> 6) & 255, h = i >> 14;
  int col = h * 64 + d;
  float sq = 0.f, sk = 0.f;
  for (int l = 0; l < 32; l++) {
    long row = (long)m * 32 + l;
    sq += qkv[row * 1536 + col];
    sk += qkv[row * 1536 + 512 + col];
  }
  ql[i] = sq * (1.f / 32.f);
  kl[i] = sk * (1.f / 32.f);
}

__global__ void k_split_cvt(const float* __restrict__ qkv, u16* __restrict__ qb,
                            u16* __restrict__ kb, u16* __restrict__ vb, long n) {
  long i = (long)blockIdx.x * blockDim.x + threadIdx.x;
  if (i < n) {
    long r = i >> 9, c = i & 511;
    qb[i] = f2b(qkv[r * 1536 + c]);
    kb[i] = f2b(qkv[r * 1536 + 512 + c]);
    vb[i] = f2b(qkv[r * 1536 + 1024 + c]);
  }
}

__global__ void k_neg_diag(const float* __restrict__ in, float* __restrict__ out,
                           float dv, long n) {
  long i = (long)blockIdx.x * blockDim.x + threadIdx.x;
  if (i < n) {
    int r = (i >> 8) & 255, c = i & 255;
    out[i] = ((r == c) ? dv : 0.f) - in[i];
  }
}

__global__ void k_init_z(const float* __restrict__ x, const float* __restrict__ scal,
                         float* __restrict__ z, long n) {
  long i = (long)blockIdx.x * blockDim.x + threadIdx.x;
  if (i < n) {
    float inv = 1.f / (scal[0] * scal[1]);
    long h = i >> 16;
    int r = (i >> 8) & 255, c = i & 255;
    z[i] = x[(h << 16) + ((long)c << 8) + r] * inv;
  }
}

__global__ void k_abssum_max(const float* __restrict__ x, int* __restrict__ scalbits,
                             int which) {
  __shared__ float red[256];
  int b = blockIdx.x, h = b >> 8, rc = b & 255, tid = threadIdx.x;
  const float* basep = x + ((long)h << 16);
  float v = (which == 0) ? fabsf(basep[(long)rc * 256 + tid])
                         : fabsf(basep[(long)tid * 256 + rc]);
  red[tid] = v; __syncthreads();
  for (int o = 128; o > 0; o >>= 1) {
    if (tid < o) red[tid] += red[tid + o];
    __syncthreads();
  }
  if (tid == 0) atomicMax(&scalbits[which], __float_as_int(red[0]));
}

__global__ void k_conv_add(const float* __restrict__ qkv, const float* __restrict__ cw,
                           float* __restrict__ outp, long n) {
  long i = (long)blockIdx.x * blockDim.x + threadIdx.x;
  if (i >= n) return;
  long row = i >> 9; int c = i & 511, h = c >> 6;
  float s = 0.f;
  #pragma unroll
  for (int t = 0; t < CONVK; t++) {
    long nn = row + t - (CONVK / 2);
    if (nn >= 0 && nn < NSEQ) s += cw[h * CONVK + t] * qkv[nn * 1536 + 1024 + c];
  }
  outp[i] += s;
}

__global__ void k_add_cvt(const float* __restrict__ a, const float* __restrict__ b,
                          float* __restrict__ enc, u16* __restrict__ encb, long n) {
  long i = (long)blockIdx.x * blockDim.x + threadIdx.x;
  if (i < n) { float s = a[i] + b[i]; enc[i] = s; encb[i] = f2b(s); }
}

__global__ void k_final(const float* __restrict__ alpha, const float* __restrict__ value,
                        const float* __restrict__ enc, float* __restrict__ xo, long n) {
  long i = (long)blockIdx.x * blockDim.x + threadIdx.x;
  if (i < n) {
    long row = i >> 9;
    float xl = alpha[row] * value[i];
    float w = 1.f / (1.f + expf(xl));   // sigmoid(-xl)
    float sq = w * w;
    xo[i] = xl * 2.f * sq + 2.f * enc[i] * (1.f - sq);
  }
}

// ---------- softmax kernels ----------
__global__ void k_row_softmax(const float* __restrict__ in, float* __restrict__ outf,
                              u16* __restrict__ outb, int cols) {
  __shared__ float red[256];
  long row = blockIdx.x;
  const float* x = in + row * (long)cols;
  int tid = threadIdx.x;
  float m = -3.4e38f;
  for (int c = tid; c < cols; c += 256) m = fmaxf(m, x[c]);
  red[tid] = m; __syncthreads();
  for (int o = 128; o > 0; o >>= 1) { if (tid < o) red[tid] = fmaxf(red[tid], red[tid + o]); __syncthreads(); }
  float bm = red[0]; __syncthreads();
  float s = 0.f;
  for (int c = tid; c < cols; c += 256) s += expf(x[c] - bm);
  red[tid] = s; __syncthreads();
  for (int o = 128; o > 0; o >>= 1) { if (tid < o) red[tid] += red[tid + o]; __syncthreads(); }
  float inv = 1.f / red[0];
  for (int c = tid; c < cols; c += 256) {
    float e = expf(x[c] - bm) * inv;
    if (outf) outf[row * (long)cols + c] = e;
    if (outb) outb[row * (long)cols + c] = f2b(e);
  }
}

__global__ void k_vec_softmax(const float* __restrict__ x, float* __restrict__ y, int n) {
  __shared__ float red[1024];
  int tid = threadIdx.x;
  float m = -3.4e38f;
  for (int c = tid; c < n; c += 1024) m = fmaxf(m, x[c]);
  red[tid] = m; __syncthreads();
  for (int o = 512; o > 0; o >>= 1) { if (tid < o) red[tid] = fmaxf(red[tid], red[tid + o]); __syncthreads(); }
  float bm = red[0]; __syncthreads();
  float s = 0.f;
  for (int c = tid; c < n; c += 1024) s += expf(x[c] - bm);
  red[tid] = s; __syncthreads();
  for (int o = 512; o > 0; o >>= 1) { if (tid < o) red[tid] += red[tid + o]; __syncthreads(); }
  float inv = 1.f / red[0];
  for (int c = tid; c < n; c += 1024) y[c] = expf(x[c] - bm) * inv;
}

// ---------- bf16 WMMA GEMM: C = alpha * A(bf16) @ B(bf16) (+bias) ----------
// 64x64 block tile, BK=32, 128 threads (4 waves), each wave a 32x32 sub-tile of
// v_wmma_f32_16x16x32_bf16. A tile fetched with ASYNC global->LDS b128 loads;
// B tile (non-transposed case) fetched by the Tensor Data Mover with hardware
// LDS row padding (interval 32 DWORDs, pad 4 DWORDs -> 144B LDS row stride).
template <bool TRANS_B>
__global__ void __launch_bounds__(128)
gemm_bf16_kernel(const u16* __restrict__ A, long lda, long sA,
                 const u16* __restrict__ B, long ldb, long sB,
                 float* __restrict__ C, long ldc, long sC,
                 int M, int N, int K, const float* __restrict__ bias, float alpha) {
  __shared__ __align__(16) u16 As[64][40];   // 80B rows (16B aligned)
  __shared__ __align__(16) u16 Bs[32][72];   // 144B rows (16B aligned)
  int tid = threadIdx.x;
  int wave = tid >> 5, lane = tid & 31, half = lane >> 4, mn = lane & 15;
  int wr = (wave & 1) * 32, wc = (wave >> 1) * 32;
  const u16* Ab = A + (long)blockIdx.z * sA;
  const u16* Bb = B + (long)blockIdx.z * sB;
  float*     Cb = C + (long)blockIdx.z * sC;
  int row0 = blockIdx.y * 64, col0 = blockIdx.x * 64;

  v8f acc[2][2];
  #pragma unroll
  for (int i = 0; i < 2; i++)
    #pragma unroll
    for (int j = 0; j < 2; j++) acc[i][j] = (v8f){};

  for (int k0 = 0; k0 < K; k0 += 32) {
    // ---- B tile ----
    if (!TRANS_B) {
#if HAS_TDM
      if (wave == 0)
        tdm_load_2d_bf16((u32)(unsigned long long)&Bs[0][0],
                         Bb + (long)k0 * ldb + col0, 64, 32, ldb,
                         /*interval 32 DWORDs*/4, /*pad 4 DWORDs*/3);
#elif HAS_ASYNC_LDS
      for (int i = tid; i < 256; i += 128) {
        int r = i >> 3, c8 = (i & 7) * 8;
        async_ldg_b128(&Bb[(long)(k0 + r) * ldb + col0 + c8], &Bs[r][c8]);
      }
#else
      for (int i = tid; i < 1024; i += 128) {
        int r = i >> 5, c2 = (i & 31) * 2;
        *(u32*)&Bs[r][c2] = *(const u32*)&Bb[(long)(k0 + r) * ldb + col0 + c2];
      }
#endif
    } else {
      for (int i = tid; i < 2048; i += 128) {
        int nn = i >> 5, kk = i & 31;
        Bs[kk][nn] = Bb[(long)(col0 + nn) * ldb + k0 + kk];
      }
    }
    // ---- A tile (async b128 when available) ----
#if HAS_ASYNC_LDS
    for (int i = tid; i < 256; i += 128) {
      int r = i >> 2, c8 = (i & 3) * 8;
      async_ldg_b128(&Ab[(long)(row0 + r) * lda + k0 + c8], &As[r][c8]);
    }
    wait_async0();
#else
    for (int i = tid; i < 1024; i += 128) {
      int r = i >> 4, c2 = (i & 15) * 2;
      *(u32*)&As[r][c2] = *(const u32*)&Ab[(long)(row0 + r) * lda + k0 + c2];
    }
#endif
#if HAS_TDM
    if (!TRANS_B && wave == 0) wait_tensor0();
#endif
    __syncthreads();

    Frag16 a[2], b[2];
    #pragma unroll
    for (int t = 0; t < 2; t++) {
      int ar = wr + t * 16 + mn;
      int bc = wc + t * 16 + mn;
      #pragma unroll
      for (int e = 0; e < 16; e++) {
        int kk2 = ((e >> 3) << 4) + (half << 3) + (e & 7);
        a[t].u[e] = As[ar][kk2];
        b[t].u[e] = Bs[kk2][bc];
      }
    }
    #pragma unroll
    for (int i = 0; i < 2; i++)
      #pragma unroll
      for (int j = 0; j < 2; j++)
        acc[i][j] = __builtin_amdgcn_wmma_f32_16x16x32_bf16(
            false, a[i].v, false, b[j].v, (short)0, acc[i][j], false, false);
    __syncthreads();
  }

  #pragma unroll
  for (int i = 0; i < 2; i++)
    #pragma unroll
    for (int j = 0; j < 2; j++) {
      int gn = col0 + wc + j * 16 + mn;
      #pragma unroll
      for (int r = 0; r < 8; r++) {
        int gm = row0 + wr + i * 16 + half * 8 + r;
        float v = acc[i][j][r] * alpha;
        if (bias) v += bias[gn];
        Cb[(long)gm * ldc + gn] = v;
      }
    }
}

// ---------- f32 WMMA GEMM (Newton-Schulz pinv): v_wmma_f32_16x16x4_f32 ----------
__global__ void __launch_bounds__(128)
gemm_f32_kernel(const float* __restrict__ A, long lda, long sA,
                const float* __restrict__ B, long ldb, long sB,
                float* __restrict__ C, long ldc, long sC,
                int M, int N, int K, float alpha) {
  __shared__ float As[64][17];
  __shared__ float Bs[16][65];
  int tid = threadIdx.x;
  int wave = tid >> 5, lane = tid & 31, half = lane >> 4, mn = lane & 15;
  int wr = (wave & 1) * 32, wc = (wave >> 1) * 32;
  const float* Ab = A + (long)blockIdx.z * sA;
  const float* Bb = B + (long)blockIdx.z * sB;
  float*       Cb = C + (long)blockIdx.z * sC;
  int row0 = blockIdx.y * 64, col0 = blockIdx.x * 64;

  v8f acc[2][2];
  #pragma unroll
  for (int i = 0; i < 2; i++)
    #pragma unroll
    for (int j = 0; j < 2; j++) acc[i][j] = (v8f){};

  for (int k0 = 0; k0 < K; k0 += 16) {
    for (int i = tid; i < 1024; i += 128) {
      int r = i >> 4, c = i & 15;
      As[r][c] = Ab[(long)(row0 + r) * lda + k0 + c];
    }
    for (int i = tid; i < 1024; i += 128) {
      int r = i >> 6, c = i & 63;
      Bs[r][c] = Bb[(long)(k0 + r) * ldb + col0 + c];
    }
    __syncthreads();
    #pragma unroll
    for (int kk = 0; kk < 4; kk++) {
      v2f a[2], b[2];
      #pragma unroll
      for (int t = 0; t < 2; t++) {
        int m = wr + t * 16 + mn;
        int nn = wc + t * 16 + mn;
        a[t][0] = As[m][kk * 4 + half * 2 + 0];
        a[t][1] = As[m][kk * 4 + half * 2 + 1];
        b[t][0] = Bs[kk * 4 + half * 2 + 0][nn];
        b[t][1] = Bs[kk * 4 + half * 2 + 1][nn];
      }
      #pragma unroll
      for (int i = 0; i < 2; i++)
        #pragma unroll
        for (int j = 0; j < 2; j++)
          acc[i][j] = __builtin_amdgcn_wmma_f32_16x16x4_f32(
              false, a[i], false, b[j], (short)0, acc[i][j], false, false);
    }
    __syncthreads();
  }

  #pragma unroll
  for (int i = 0; i < 2; i++)
    #pragma unroll
    for (int j = 0; j < 2; j++) {
      int gn = col0 + wc + j * 16 + mn;
      #pragma unroll
      for (int r = 0; r < 8; r++) {
        int gm = row0 + wr + i * 16 + half * 8 + r;
        Cb[(long)gm * ldc + gn] = acc[i][j][r] * alpha;
      }
    }
}

// ---------- fused masked score row-sum: A_raw[i] += scale * sum_j adj[i,j]*(q_i.k_j) ----
// 128x128 tile per block, 8 waves; the 8192x8192 score matrix is never
// materialized. adj (268 MB, the roofline-dominant stream) is read exactly once,
// with global_prefetch issued before the WMMA K-loop to hide HBM latency.
__global__ void __launch_bounds__(256)
masked_rowsum_kernel(const u16* __restrict__ qb, const u16* __restrict__ kb,
                     const float* __restrict__ adj, float* __restrict__ Araw,
                     int Nn, float scale) {
  __shared__ __align__(16) u16 As[128][40];
  __shared__ __align__(16) u16 Bs[32][130];
  int tid = threadIdx.x;
  int wave = tid >> 5, lane = tid & 31, half = lane >> 4, mn = lane & 15;
  int rb = blockIdx.y * 128, cb = blockIdx.x * 128;
  int gm0 = rb + 16 * wave + 8 * half;

  // prefetch the adj tile consumed after the K-loop (-> global_prefetch_b8)
  #pragma unroll
  for (int r = 0; r < 8; r++)
    __builtin_prefetch(&adj[(long)(gm0 + r) * Nn + cb + mn * 8], 0, 1);

  v8f acc[8];
  #pragma unroll
  for (int t = 0; t < 8; t++) acc[t] = (v8f){};

  for (int k0 = 0; k0 < ATTD; k0 += 32) {
#if HAS_ASYNC_LDS
    for (int i = tid; i < 512; i += 256) {
      int r = i >> 2, c8 = (i & 3) * 8;
      async_ldg_b128(&qb[(long)(rb + r) * ATTD + k0 + c8], &As[r][c8]);
    }
#else
    for (int i = tid; i < 2048; i += 256) {
      int r = i >> 4, c2 = (i & 15) * 2;
      *(u32*)&As[r][c2] = *(const u32*)&qb[(long)(rb + r) * ATTD + k0 + c2];
    }
#endif
    for (int i = tid; i < 4096; i += 256) {
      int nn = i >> 5, kk = i & 31;
      Bs[kk][nn] = kb[(long)(cb + nn) * ATTD + k0 + kk];
    }
#if HAS_ASYNC_LDS
    wait_async0();
#endif
    __syncthreads();

    Frag16 a;
    int ar = 16 * wave + mn;
    #pragma unroll
    for (int e = 0; e < 16; e++) {
      int kk2 = ((e >> 3) << 4) + (half << 3) + (e & 7);
      a.u[e] = As[ar][kk2];
    }
    #pragma unroll
    for (int ct = 0; ct < 8; ct++) {
      Frag16 b;
      #pragma unroll
      for (int e = 0; e < 16; e++) {
        int kk2 = ((e >> 3) << 4) + (half << 3) + (e & 7);
        b.u[e] = Bs[kk2][ct * 16 + mn];
      }
      acc[ct] = __builtin_amdgcn_wmma_f32_16x16x32_bf16(
          false, a.v, false, b.v, (short)0, acc[ct], false, false);
    }
    __syncthreads();
  }

  float rs[8];
  #pragma unroll
  for (int r = 0; r < 8; r++) rs[r] = 0.f;
  #pragma unroll
  for (int ct = 0; ct < 8; ct++) {
    int gn = cb + ct * 16 + mn;
    #pragma unroll
    for (int r = 0; r < 8; r++)
      rs[r] += acc[ct][r] * adj[(long)(gm0 + r) * Nn + gn];
  }
  #pragma unroll
  for (int r = 0; r < 8; r++) {
    rs[r] += __shfl_xor(rs[r], 1, 16);
    rs[r] += __shfl_xor(rs[r], 2, 16);
    rs[r] += __shfl_xor(rs[r], 4, 16);
    rs[r] += __shfl_xor(rs[r], 8, 16);
  }
  if (mn == 0) {
    #pragma unroll
    for (int r = 0; r < 8; r++) atomicAdd(&Araw[gm0 + r], rs[r] * scale);
  }
}

// ---------- host orchestration ----------
static void launch_gemm_bf16(bool transB, const u16* A, long lda, long sA,
                             const u16* B, long ldb, long sB,
                             float* C, long ldc, long sC,
                             int M, int N, int K, int batch,
                             const float* bias, float alpha, hipStream_t s) {
  dim3 g(N / 64, M / 64, batch), b(128);
  if (transB)
    gemm_bf16_kernel<true><<<g, b, 0, s>>>(A, lda, sA, B, ldb, sB, C, ldc, sC, M, N, K, bias, alpha);
  else
    gemm_bf16_kernel<false><<<g, b, 0, s>>>(A, lda, sA, B, ldb, sB, C, ldc, sC, M, N, K, bias, alpha);
}

static void launch_gemm_f32(const float* A, long lda, long sA, const float* B, long ldb, long sB,
                            float* C, long ldc, long sC, int M, int N, int K, int batch,
                            float alpha, hipStream_t s) {
  dim3 g(N / 64, M / 64, batch), b(128);
  gemm_f32_kernel<<<g, b, 0, s>>>(A, lda, sA, B, ldb, sB, C, ldc, sC, M, N, K, alpha);
}

extern "C" void kernel_launch(void* const* d_in, const int* in_sizes, int n_in,
                              void* d_out, int out_size, void* d_ws, size_t ws_size,
                              hipStream_t stream) {
  const float* dense  = (const float*)d_in[0];
  const float* adj    = (const float*)d_in[1];
  const float* wq     = (const float*)d_in[2];
  const float* wk     = (const float*)d_in[3];
  const float* wv_w   = (const float*)d_in[4];
  const float* wv_b   = (const float*)d_in[5];
  const float* qkv_w  = (const float*)d_in[6];
  const float* out_w  = (const float*)d_in[7];
  const float* out_b  = (const float*)d_in[8];
  const float* conv_w = (const float*)d_in[9];

  char* base = (char*)d_ws;
  size_t off = 0;
  auto alloc = [&](size_t bytes) -> void* {
    off = (off + 255) & ~(size_t)255;
    void* p = base + off;
    off += bytes;
    return p;
  };
  const long ND = (long)NSEQ * DMODEL;

  u16* Xb     = (u16*)alloc(ND * 2);
  u16* Wqkvb  = (u16*)alloc((size_t)DMODEL * 1536 * 2);
  u16* Woutb  = (u16*)alloc((size_t)DMODEL * DMODEL * 2);
  u16* Wqb    = (u16*)alloc((size_t)DMODEL * ATTD * 2);
  u16* Wkb    = (u16*)alloc((size_t)DMODEL * ATTD * 2);
  u16* Wvb    = (u16*)alloc((size_t)DMODEL * DMODEL * 2);
  float* QKV  = (float*)alloc((size_t)NSEQ * 1536 * 4);
  u16* qb     = (u16*)alloc(ND * 2);
  u16* kb     = (u16*)alloc(ND * 2);
  u16* vb     = (u16*)alloc(ND * 2);
  long LHD    = (long)NHEAD * NLAND * DHEAD;
  float* ql   = (float*)alloc(LHD * 4);
  float* kl   = (float*)alloc(LHD * 4);
  u16* qlb    = (u16*)alloc(LHD * 2);
  u16* klb    = (u16*)alloc(LHD * 2);
  long A1     = (long)NHEAD * NSEQ * NLAND;
  float* l1   = (float*)alloc(A1 * 4);
  u16* a1b    = (u16*)alloc(A1 * 2);
  long A2     = (long)NHEAD * NLAND * NLAND;
  float* l2   = (float*)alloc(A2 * 4);
  float* a2f  = (float*)alloc(A2 * 4);
  long A3     = (long)NHEAD * NLAND * NSEQ;
  float* l3   = (float*)alloc(A3 * 4);
  u16* a3b    = (u16*)alloc(A3 * 2);
  float* z0   = (float*)alloc(A2 * 4);
  float* z1   = (float*)alloc(A2 * 4);
  float* ta   = (float*)alloc(A2 * 4);
  float* tb   = (float*)alloc(A2 * 4);
  float* tc   = (float*)alloc(A2 * 4);
  float* scal = (float*)alloc(256);
  u16* a2invb = (u16*)alloc(A2 * 2);
  float* a3v  = (float*)alloc(LHD * 4);
  u16* a3vb   = (u16*)alloc(LHD * 2);
  float* t2   = (float*)alloc(LHD * 4);
  u16* t2b    = (u16*)alloc(LHD * 2);
  float* aout = (float*)alloc(ND * 4);
  u16* aob    = (u16*)alloc(ND * 2);
  float* xg   = (float*)alloc(ND * 4);
  float* enc  = (float*)alloc(ND * 4);
  u16* encb   = (u16*)alloc(ND * 2);
  long NA     = (long)NSEQ * ATTD;
  float* qc   = (float*)alloc(NA * 4);
  float* kc   = (float*)alloc(NA * 4);
  u16* qcb    = (u16*)alloc(NA * 2);
  u16* kcb    = (u16*)alloc(NA * 2);
  float* alphaBuf = (float*)alloc((size_t)NSEQ * 4);
  float* value    = (float*)alloc(ND * 4);

  float* xo   = (float*)d_out;
  float* Araw = (float*)d_out + ND;

  auto eb = [](long n) { return dim3((unsigned)((n + 255) / 256)); };

  // 1. bf16 conversions of inputs
  k_cvt_bf16<<<eb(ND), 256, 0, stream>>>(dense, Xb, ND);
  k_cvt_bf16<<<eb((long)DMODEL * 1536), 256, 0, stream>>>(qkv_w, Wqkvb, (long)DMODEL * 1536);
  k_cvt_bf16<<<eb((long)DMODEL * DMODEL), 256, 0, stream>>>(out_w, Woutb, (long)DMODEL * DMODEL);
  k_cvt_bf16<<<eb((long)DMODEL * ATTD), 256, 0, stream>>>(wq, Wqb, (long)DMODEL * ATTD);
  k_cvt_bf16<<<eb((long)DMODEL * ATTD), 256, 0, stream>>>(wk, Wkb, (long)DMODEL * ATTD);
  k_cvt_bf16<<<eb((long)DMODEL * DMODEL), 256, 0, stream>>>(wv_w, Wvb, (long)DMODEL * DMODEL);

  // 2. QKV = X @ qkv_w   (8192x512x1536)
  launch_gemm_bf16(false, Xb, DMODEL, 0, Wqkvb, 1536, 0, QKV, 1536, 0,
                   NSEQ, 1536, DMODEL, 1, nullptr, 1.f, stream);
  // 3. q *= dh^-0.5 ; landmarks ; split/convert
  k_scale_q<<<eb(ND), 256, 0, stream>>>(QKV, 0.125f, ND);
  k_landmark_mean<<<eb(LHD), 256, 0, stream>>>(QKV, ql, kl);
  k_cvt_bf16<<<eb(LHD), 256, 0, stream>>>(ql, qlb, LHD);
  k_cvt_bf16<<<eb(LHD), 256, 0, stream>>>(kl, klb, LHD);
  k_split_cvt<<<eb(ND), 256, 0, stream>>>(QKV, qb, kb, vb, ND);

  // 4. attention logits (batched per head)
  launch_gemm_bf16(true, qb, DMODEL, DHEAD, klb, DHEAD, (long)NLAND * DHEAD,
                   l1, NLAND, (long)NSEQ * NLAND, NSEQ, NLAND, DHEAD, NHEAD, nullptr, 1.f, stream);
  launch_gemm_bf16(true, qlb, DHEAD, (long)NLAND * DHEAD, klb, DHEAD, (long)NLAND * DHEAD,
                   l2, NLAND, (long)NLAND * NLAND, NLAND, NLAND, DHEAD, NHEAD, nullptr, 1.f, stream);
  launch_gemm_bf16(true, qlb, DHEAD, (long)NLAND * DHEAD, kb, DMODEL, DHEAD,
                   l3, NSEQ, (long)NLAND * NSEQ, NLAND, NSEQ, DHEAD, NHEAD, nullptr, 1.f, stream);

  // 5. softmaxes
  k_row_softmax<<<dim3((unsigned)(NHEAD * NSEQ)), 256, 0, stream>>>(l1, nullptr, a1b, NLAND);
  k_row_softmax<<<dim3((unsigned)(NHEAD * NLAND)), 256, 0, stream>>>(l2, a2f, nullptr, NLAND);
  k_row_softmax<<<dim3((unsigned)(NHEAD * NLAND)), 256, 0, stream>>>(l3, nullptr, a3b, NSEQ);

  // 6. Moore-Penrose pinv of attn2 (f32 WMMA), 6 Newton-Schulz iterations
  (void)hipMemsetAsync(scal, 0, 8, stream);
  k_abssum_max<<<dim3(NHEAD * NLAND), 256, 0, stream>>>(a2f, (int*)scal, 0);
  k_abssum_max<<<dim3(NHEAD * NLAND), 256, 0, stream>>>(a2f, (int*)scal, 1);
  k_init_z<<<eb(A2), 256, 0, stream>>>(a2f, scal, z0, A2);
  float* zc = z0; float* zn = z1;
  for (int it = 0; it < 6; it++) {
    launch_gemm_f32(a2f, NLAND, (long)NLAND * NLAND, zc, NLAND, (long)NLAND * NLAND,
                    ta, NLAND, (long)NLAND * NLAND, NLAND, NLAND, NLAND, NHEAD, 1.f, stream);
    k_neg_diag<<<eb(A2), 256, 0, stream>>>(ta, tb, 7.f, A2);
    launch_gemm_f32(ta, NLAND, (long)NLAND * NLAND, tb, NLAND, (long)NLAND * NLAND,
                    tc, NLAND, (long)NLAND * NLAND, NLAND, NLAND, NLAND, NHEAD, 1.f, stream);
    k_neg_diag<<<eb(A2), 256, 0, stream>>>(tc, tb, 15.f, A2);
    launch_gemm_f32(ta, NLAND, (long)NLAND * NLAND, tb, NLAND, (long)NLAND * NLAND,
                    tc, NLAND, (long)NLAND * NLAND, NLAND, NLAND, NLAND, NHEAD, 1.f, stream);
    k_neg_diag<<<eb(A2), 256, 0, stream>>>(tc, tb, 13.f, A2);
    launch_gemm_f32(zc, NLAND, (long)NLAND * NLAND, tb, NLAND, (long)NLAND * NLAND,
                    zn, NLAND, (long)NLAND * NLAND, NLAND, NLAND, NLAND, NHEAD, 0.25f, stream);
    float* t = zc; zc = zn; zn = t;
  }
  k_cvt_bf16<<<eb(A2), 256, 0, stream>>>(zc, a2invb, A2);

  // 7. out = attn1 @ (attn2_inv @ (attn3 @ v))
  launch_gemm_bf16(false, a3b, NSEQ, (long)NLAND * NSEQ, vb, DMODEL, DHEAD,
                   a3v, DHEAD, (long)NLAND * DHEAD, NLAND, DHEAD, NSEQ, NHEAD, nullptr, 1.f, stream);
  k_cvt_bf16<<<eb(LHD), 256, 0, stream>>>(a3v, a3vb, LHD);
  launch_gemm_bf16(false, a2invb, NLAND, (long)NLAND * NLAND, a3vb, DHEAD, (long)NLAND * DHEAD,
                   t2, DHEAD, (long)NLAND * DHEAD, NLAND, DHEAD, NLAND, NHEAD, nullptr, 1.f, stream);
  k_cvt_bf16<<<eb(LHD), 256, 0, stream>>>(t2, t2b, LHD);
  launch_gemm_bf16(false, a1b, NLAND, (long)NSEQ * NLAND, t2b, DHEAD, (long)NLAND * DHEAD,
                   aout, DMODEL, DHEAD, NSEQ, DHEAD, NLAND, NHEAD, nullptr, 1.f, stream);

  // 8. depthwise residual conv (reads f32 v inside QKV) added into aout
  k_conv_add<<<eb(ND), 256, 0, stream>>>(QKV, conv_w, aout, ND);

  // 9. output projection + residual -> enc
  k_cvt_bf16<<<eb(ND), 256, 0, stream>>>(aout, aob, ND);
  launch_gemm_bf16(false, aob, DMODEL, 0, Woutb, DMODEL, 0, xg, DMODEL, 0,
                   NSEQ, DMODEL, DMODEL, 1, out_b, 1.f, stream);
  k_add_cvt<<<eb(ND), 256, 0, stream>>>(xg, dense, enc, encb, ND);

  // 10. custom attention: qc = enc@wq, kc = enc@wk
  launch_gemm_bf16(false, encb, DMODEL, 0, Wqb, ATTD, 0, qc, ATTD, 0,
                   NSEQ, ATTD, DMODEL, 1, nullptr, 1.f, stream);
  launch_gemm_bf16(false, encb, DMODEL, 0, Wkb, ATTD, 0, kc, ATTD, 0,
                   NSEQ, ATTD, DMODEL, 1, nullptr, 1.f, stream);
  k_cvt_bf16<<<eb(NA), 256, 0, stream>>>(qc, qcb, NA);
  k_cvt_bf16<<<eb(NA), 256, 0, stream>>>(kc, kcb, NA);

  // 11. A_raw = rowsum(adj * (qc@kc^T)/16)  — fused, scores never materialized
  (void)hipMemsetAsync(Araw, 0, (size_t)NSEQ * 4, stream);
  masked_rowsum_kernel<<<dim3(NSEQ / 128, NSEQ / 128), 256, 0, stream>>>(
      qcb, kcb, adj, Araw, NSEQ, 1.f / 16.f);

  // 12. alpha = softmax(A_raw); value = dense@wv_w + wv_b; fused epilogue
  k_vec_softmax<<<1, 1024, 0, stream>>>(Araw, alphaBuf, NSEQ);
  launch_gemm_bf16(false, Xb, DMODEL, 0, Wvb, DMODEL, 0, value, DMODEL, 0,
                   NSEQ, DMODEL, DMODEL, 1, wv_b, 1.f, stream);
  k_final<<<eb(ND), 256, 0, stream>>>(alphaBuf, value, enc, xo, ND);

  (void)in_sizes; (void)n_in; (void)out_size; (void)ws_size;
}
